// Net_22557168238620
// MI455X (gfx1250) — matrix operand order repaired and checked by reading, not compiled, not run on hardware
//
#include <hip/hip_runtime.h>
#include <hip/hip_bf16.h>

typedef __attribute__((ext_vector_type(16))) _Float16 v16h;
typedef __attribute__((ext_vector_type(8)))  float    v8f;

#define NFEAT  512
#define NHID   16
#define NCLASS 40

// ---------------------------------------------------------------- degree prep
__global__ void k_init_deg(float* dinv, int n) {
  int i = blockIdx.x * blockDim.x + threadIdx.x;
  if (i < n) dinv[i] = 1.0f;                     // self loop -> deg >= 1
}

__global__ void k_deg(const long long* __restrict__ src, float* dinv, int E) {
  int e = blockIdx.x * blockDim.x + threadIdx.x;
  if (e < E) atomicAdd(&dinv[(int)src[e]], 1.0f);
}

__global__ void k_rsqrt(float* dinv, int n) {
  int i = blockIdx.x * blockDim.x + threadIdx.x;
  if (i < n) dinv[i] = rsqrtf(dinv[i]);
}

// ---------------------------------------------------------------- layer-1 GEMM
// h1[n,16] = x[n,512] @ W1[512,16] + b1, f16 WMMA with f32 accumulation.
// 256 threads = 8 wave32; each wave computes one 16(M) x 16(N) tile, K in 16
// steps of 32 via v_wmma_f32_16x16x32_f16.
__global__ void __launch_bounds__(256)
k_gemm1_wmma(const float* __restrict__ x, const float* __restrict__ W1,
             const float* __restrict__ b1, float* __restrict__ h1, int n) {
  __shared__ _Float16 lW[NHID * NFEAT];          // transposed: lW[col][k], 16KB
  __shared__ float    lB[NHID];

  for (int idx = threadIdx.x; idx < NHID * NFEAT; idx += blockDim.x) {
    int col = idx / NFEAT, k = idx % NFEAT;
    lW[idx] = (_Float16)W1[k * NHID + col];
  }
  if (threadIdx.x < NHID) lB[threadIdx.x] = b1[threadIdx.x];
  __syncthreads();

  const int wave = threadIdx.x >> 5;
  const int lane = threadIdx.x & 31;
  const int hlf  = lane >> 4;                    // half-wave selector
  const int nidx = lane & 15;                    // output column
  const int rowBase = blockIdx.x * 128 + wave * 16;

  int m  = rowBase + nidx;                       // A-fragment row for this lane
  int mc = (m < n) ? m : (n - 1);                // clamp: keep EXEC all-1s
  const float* xr = x + (size_t)mc * NFEAT;

  v8f c = {};
#pragma unroll 4
  for (int kb = 0; kb < NFEAT / 32; ++kb) {
    const int k0 = kb * 32;
    __builtin_prefetch(xr + k0 + 128, 0, 1);     // global_prefetch_b8 next tiles

    // A 16x32 f16 fragment per ISA layout (lanes 0-15: K 0-7/16-23, 16-31: +8)
    const float4* p0 = (const float4*)(xr + k0 + hlf * 8);
    const float4* p1 = (const float4*)(xr + k0 + 16 + hlf * 8);
    float4 q0 = p0[0], q1 = p0[1], q2 = p1[0], q3 = p1[1];
    v16h a;
    a[0]  = (_Float16)q0.x; a[1]  = (_Float16)q0.y;
    a[2]  = (_Float16)q0.z; a[3]  = (_Float16)q0.w;
    a[4]  = (_Float16)q1.x; a[5]  = (_Float16)q1.y;
    a[6]  = (_Float16)q1.z; a[7]  = (_Float16)q1.w;
    a[8]  = (_Float16)q2.x; a[9]  = (_Float16)q2.y;
    a[10] = (_Float16)q2.z; a[11] = (_Float16)q2.w;
    a[12] = (_Float16)q3.x; a[13] = (_Float16)q3.y;
    a[14] = (_Float16)q3.z; a[15] = (_Float16)q3.w;

    // B 32x16 fragment: lane col = nidx, lanes 0-15 K 0-7/16-23, lanes 16-31 +8
    const _Float16* bp = &lW[nidx * NFEAT + k0 + hlf * 8];
    v16h b;
#pragma unroll
    for (int i = 0; i < 16; ++i) b[i] = bp[i + (i >= 8 ? 8 : 0)];

    c = __builtin_amdgcn_wmma_f32_16x16x32_f16(false, a, false, b,
                                               (short)0, c, false, false);
  }

  // C/D layout: VGPR r -> row (rowBase + hlf*8 + r), col nidx
#pragma unroll
  for (int r = 0; r < 8; ++r) {
    int mm = rowBase + hlf * 8 + r;
    if (mm < n) h1[(size_t)mm * NHID + nidx] = c[r] + lB[nidx];
  }
}

// ---------------------------------------------------------------- aggregation 1
__global__ void k_self1(const float* __restrict__ h1, const float* __restrict__ dinv,
                        float* __restrict__ agg1, int n) {
  int t = blockIdx.x * blockDim.x + threadIdx.x;
  if (t < n * NHID) {
    int i = t / NHID;
    float di = dinv[i];
    agg1[t] = di * di * h1[t];
  }
}

__global__ void k_scatter1(const long long* __restrict__ src,
                           const long long* __restrict__ dst,
                           const float* __restrict__ h1,
                           const float* __restrict__ dinv,
                           float* __restrict__ agg1, int E) {
  int e = blockIdx.x * blockDim.x + threadIdx.x;
  if (e >= E) return;
  int s = (int)src[e], d = (int)dst[e];
  float nm = dinv[s] * dinv[d];
  const float* hs = h1 + (size_t)s * NHID;
  float* ad = agg1 + (size_t)d * NHID;
#pragma unroll
  for (int f = 0; f < NHID; ++f) atomicAdd(&ad[f], nm * hs[f]);
}

// ---------------------------------------------------------------- layer-2 GEMM
// K=16 -> trivial; one thread per (node, class), ReLU fused on the input.
__global__ void k_gemm2(const float* __restrict__ agg1, const float* __restrict__ W2,
                        const float* __restrict__ b2, float* __restrict__ h2, int n) {
  int t = blockIdx.x * blockDim.x + threadIdx.x;
  if (t >= n * NCLASS) return;
  int i = t / NCLASS, c = t % NCLASS;
  const float* ar = agg1 + (size_t)i * NHID;
  float acc = b2[c];
#pragma unroll
  for (int k = 0; k < NHID; ++k) {
    float v = ar[k];
    v = v > 0.0f ? v : 0.0f;
    acc = fmaf(v, W2[k * NCLASS + c], acc);
  }
  h2[t] = acc;
}

// ---------------------------------------------------------------- aggregation 2
__global__ void k_self2(const float* __restrict__ h2, const float* __restrict__ dinv,
                        float* __restrict__ out, int n) {
  int t = blockIdx.x * blockDim.x + threadIdx.x;
  if (t < n * NCLASS) {
    int i = t / NCLASS;
    float di = dinv[i];
    out[t] = di * di * h2[t];
  }
}

__global__ void k_scatter2(const long long* __restrict__ src,
                           const long long* __restrict__ dst,
                           const float* __restrict__ h2,
                           const float* __restrict__ dinv,
                           float* __restrict__ out, int E) {
  int e = blockIdx.x * blockDim.x + threadIdx.x;
  if (e >= E) return;
  int s = (int)src[e], d = (int)dst[e];
  float nm = dinv[s] * dinv[d];
  const float* hs = h2 + (size_t)s * NCLASS;
  float* od = out + (size_t)d * NCLASS;
#pragma unroll
  for (int c = 0; c < NCLASS; ++c) atomicAdd(&od[c], nm * hs[c]);
}

// ---------------------------------------------------------------- log_softmax
__global__ void k_logsoftmax(float* __restrict__ out, int n) {
  int i = blockIdx.x * blockDim.x + threadIdx.x;
  if (i >= n) return;
  float* r = out + (size_t)i * NCLASS;
  float m = -INFINITY;
#pragma unroll
  for (int c = 0; c < NCLASS; ++c) m = fmaxf(m, r[c]);
  float s = 0.0f;
#pragma unroll
  for (int c = 0; c < NCLASS; ++c) s += __expf(r[c] - m);
  float l = m + __logf(s);
#pragma unroll
  for (int c = 0; c < NCLASS; ++c) r[c] -= l;
}

// ---------------------------------------------------------------- launch
extern "C" void kernel_launch(void* const* d_in, const int* in_sizes, int n_in,
                              void* d_out, int out_size, void* d_ws, size_t ws_size,
                              hipStream_t stream) {
  const float*     x  = (const float*)d_in[0];
  const long long* ei = (const long long*)d_in[1];   // int64 [2, E]
  const float*     W1 = (const float*)d_in[2];
  const float*     b1 = (const float*)d_in[3];
  const float*     W2 = (const float*)d_in[4];
  const float*     b2 = (const float*)d_in[5];

  const int n = in_sizes[0] / NFEAT;
  const int E = in_sizes[1] / 2;
  const long long* src = ei;
  const long long* dst = ei + E;

  float* ws   = (float*)d_ws;
  float* dinv = ws;                                  // n
  float* h1   = dinv + n;                            // n*16
  float* agg1 = h1 + (size_t)n * NHID;               // n*16
  float* h2   = agg1 + (size_t)n * NHID;             // n*40
  float* out  = (float*)d_out;                       // n*40 (agg2 in place)

  const int B = 256;
  k_init_deg  <<<(n + B - 1) / B, B, 0, stream>>>(dinv, n);
  k_deg       <<<(E + B - 1) / B, B, 0, stream>>>(src, dinv, E);
  k_rsqrt     <<<(n + B - 1) / B, B, 0, stream>>>(dinv, n);
  k_gemm1_wmma<<<(n + 127) / 128, 256, 0, stream>>>(x, W1, b1, h1, n);
  k_self1     <<<((long long)n * NHID + B - 1) / B, B, 0, stream>>>(h1, dinv, agg1, n);
  k_scatter1  <<<(E + B - 1) / B, B, 0, stream>>>(src, dst, h1, dinv, agg1, E);
  k_gemm2     <<<((long long)n * NCLASS + B - 1) / B, B, 0, stream>>>(agg1, W2, b2, h2, n);
  k_self2     <<<((long long)n * NCLASS + B - 1) / B, B, 0, stream>>>(h2, dinv, out, n);
  k_scatter2  <<<(E + B - 1) / B, B, 0, stream>>>(src, dst, h2, dinv, out, E);
  k_logsoftmax<<<(n + B - 1) / B, B, 0, stream>>>(out, n);
}
